// GATLayer_42082089566758
// MI455X (gfx1250) — matrix operand order repaired and checked by reading, not compiled, not run on hardware
//
#include <hip/hip_runtime.h>

typedef __attribute__((ext_vector_type(16))) __bf16 v16bf;
typedef __attribute__((ext_vector_type(8)))  float  v8f;

union Bregs { unsigned int u[8]; v16bf v; };

__device__ inline unsigned int bf16_rne(float f) {
  unsigned int u = __float_as_uint(f);
  return ((u + 0x7FFFu + ((u >> 16) & 1u)) >> 16) & 0xFFFFu;
}
__device__ inline unsigned int pk_bf16(float a, float b) {
  return bf16_rne(a) | (bf16_rne(b) << 16);
}

// ---------------------------------------------------------------------------
// Pre-swizzle Wv (row-major [256 out][256 in], f32) into this kernel's WMMA B-operand
// lane layout as packed bf16 pairs, organized as two 512B half-tiles so both
// the global->LDS staging copy and the per-lane ds_load_b128 reads are
// contiguous and LDS-bank-conflict-free:
//   uint32 index = tile*256 + hf*128 + lane*4 + j      (tile = kt*16 + ot)
//   value       = pack( Wv[n][k], Wv[n][k+1] )
//   n = ot*16 + (lane&15),  v = hf*4 + j,  k = kt*32 + (lane>>4)*16 + 2*v
// ---------------------------------------------------------------------------
__global__ void k_prep_w(const float* __restrict__ Wv, unsigned int* __restrict__ Wsw) {
  int gid = blockIdx.x * blockDim.x + threadIdx.x;       // 32768 total
  if (gid >= 8 * 16 * 32 * 8) return;
  int j    = gid & 3;
  int lane = (gid >> 2) & 31;
  int hf   = (gid >> 7) & 1;
  int tile = gid >> 8;                                   // 0..127
  int kt = tile >> 4, ot = tile & 15;
  int v = hf * 4 + j;
  int n = ot * 16 + (lane & 15);
  int k = kt * 32 + (lane >> 4) * 16 + 2 * v;
  Wsw[gid] = pk_bf16(Wv[n * 256 + k], Wv[n * 256 + k + 1]);
}

// ---------------------------------------------------------------------------
// h = x @ Wv.T + bv   via v_wmma_f32_16x16x32_bf16.
// One wave computes a 16-row x 256-col slab (16 accumulator tiles).
// B is staged per-kt-slice (16 KB) into double-buffered LDS; the global
// prefetch of slice kt+1 issues before the 16 WMMAs of slice kt, so the
// L2 latency is hidden behind matrix work. DS loads for tile ot+1 issue
// behind the WMMA of tile ot (one-stage pipeline).
// ---------------------------------------------------------------------------
__global__ __launch_bounds__(256) void k_gemm_h(const float* __restrict__ x,
                                                const unsigned int* __restrict__ Wsw,
                                                const float* __restrict__ bv,
                                                float* __restrict__ h,
                                                int numTiles, int N) {
  __shared__ __align__(16) unsigned int sB[2][4096];     // 2 x 16 KB kt-slices
  int tid  = threadIdx.x;
  int wave = (int)((blockIdx.x * blockDim.x + tid) >> 5);
  int lane = tid & 31;
  int lmod = lane & 15;
  int lhi  = lane >> 4;                                  // 0 or 1
  int m0   = wave * 16;

  // stage kt=0 slice into buffer 0 (each thread copies 4 x 16B, coalesced)
  {
    const uint4* g = (const uint4*)Wsw;
    uint4* s = (uint4*)&sB[0][0];
#pragma unroll
    for (int i = 0; i < 4; ++i) s[tid + 256 * i] = g[tid + 256 * i];
  }
  __syncthreads();

  v8f acc[16];
#pragma unroll
  for (int ot = 0; ot < 16; ++ot) {
    float b = bv[ot * 16 + lmod];                        // bias broadcast over M
    acc[ot] = (v8f){b, b, b, b, b, b, b, b};
  }

  int arow = m0 + lmod;
  if (arow >= N) arow = N - 1;                           // clamp (stores guarded)

#pragma unroll 1
  for (int kt = 0; kt < 8; ++kt) {
    int cur = kt & 1;

    // A operand: 16x32 bf16. lane<16: K = kt*32 + {0..15};
    //            lane>=16:   K = kt*32 + {16..31}
    const float* xr = x + (size_t)arow * 256 + kt * 32 + lhi * 16;
    float4 f0 = *(const float4*)(xr);
    float4 f1 = *(const float4*)(xr + 4);
    float4 f2 = *(const float4*)(xr + 8);
    float4 f3 = *(const float4*)(xr + 12);

    // issue the global prefetch of the NEXT kt-slice before the compute so
    // its wait lands after 16 WMMAs
    uint4 pf[4];
    if (kt < 7) {
      const uint4* g = (const uint4*)(Wsw + (kt + 1) * 4096);
#pragma unroll
      for (int i = 0; i < 4; ++i) pf[i] = g[tid + 256 * i];
    }

    // native packed f32->bf16 conversion for the A operand
    v16bf Av;
    Av[0]  = (__bf16)f0.x; Av[1]  = (__bf16)f0.y;
    Av[2]  = (__bf16)f0.z; Av[3]  = (__bf16)f0.w;
    Av[4]  = (__bf16)f1.x; Av[5]  = (__bf16)f1.y;
    Av[6]  = (__bf16)f1.z; Av[7]  = (__bf16)f1.w;
    Av[8]  = (__bf16)f2.x; Av[9]  = (__bf16)f2.y;
    Av[10] = (__bf16)f2.z; Av[11] = (__bf16)f2.w;
    Av[12] = (__bf16)f3.x; Av[13] = (__bf16)f3.y;
    Av[14] = (__bf16)f3.z; Av[15] = (__bf16)f3.w;

    // 16 WMMAs; B tiles stream from LDS (uint4 idx: ot*64 + hf*32 + lane)
    const uint4* lb = (const uint4*)&sB[cur][0];
    uint4 bA = lb[lane];                                 // ot=0 half 0
    uint4 bB = lb[32 + lane];                            // ot=0 half 1
#pragma unroll
    for (int ot = 0; ot < 16; ++ot) {
      uint4 nA, nB;
      if (ot < 15) {
        nA = lb[(ot + 1) * 64 + lane];
        nB = lb[(ot + 1) * 64 + 32 + lane];
      }
      Bregs B;
      B.u[0] = bA.x; B.u[1] = bA.y; B.u[2] = bA.z; B.u[3] = bA.w;
      B.u[4] = bB.x; B.u[5] = bB.y; B.u[6] = bB.z; B.u[7] = bB.w;
      acc[ot] = __builtin_amdgcn_wmma_f32_16x16x32_bf16(
          false, Av, false, B.v, (short)0, acc[ot], false, false);
      bA = nA; bB = nB;
    }

    // fill the other LDS buffer for kt+1; barrier also protects reuse
    if (kt < 7) {
      uint4* s = (uint4*)&sB[cur ^ 1][0];
#pragma unroll
      for (int i = 0; i < 4; ++i) s[tid + 256 * i] = pf[i];
      __syncthreads();
    }
  }

  // C layout: VGPR r -> row m0 + r (+8 for lanes 16-31), col = ot*16 + lmod
  int mlo = m0 + lhi * 8;
#pragma unroll
  for (int ot = 0; ot < 16; ++ot) {
#pragma unroll
    for (int r = 0; r < 8; ++r) {
      int row = mlo + r;
      if (row < N) h[(size_t)row * 256 + ot * 16 + lmod] = acc[ot][r];
    }
  }
}

// ---------------------------------------------------------------------------
// q = h @ Wq.T + bq ; k = h @ Wk.T + bk   (per-head scalars, wave per node)
// ---------------------------------------------------------------------------
__global__ __launch_bounds__(256) void k_qk(const float* __restrict__ h,
                                            const float* __restrict__ Wq,
                                            const float* __restrict__ bq,
                                            const float* __restrict__ Wk,
                                            const float* __restrict__ bk,
                                            float* __restrict__ q,
                                            float* __restrict__ kk, int N) {
  int wave = (int)((blockIdx.x * blockDim.x + threadIdx.x) >> 5);
  int lane = threadIdx.x & 31;
  if (wave >= N) return;
  const float* hr = h + (size_t)wave * 256;
  float hv[8];
#pragma unroll
  for (int j = 0; j < 8; ++j) hv[j] = hr[j * 32 + lane];
#pragma unroll
  for (int hd = 0; hd < 8; ++hd) {
    float sq = 0.f, sk = 0.f;
#pragma unroll
    for (int j = 0; j < 8; ++j) {
      sq += hv[j] * Wq[hd * 256 + j * 32 + lane];
      sk += hv[j] * Wk[hd * 256 + j * 32 + lane];
    }
#pragma unroll
    for (int off = 16; off > 0; off >>= 1) {
      sq += __shfl_down(sq, off, 32);
      sk += __shfl_down(sk, off, 32);
    }
    if (lane == 0) {
      q [wave * 8 + hd] = sq + bq[hd];
      kk[wave * 8 + hd] = sk + bk[hd];
    }
  }
}

// --------------------------- CSR construction ------------------------------
__global__ void k_zero_i32(int* p, int n) {
  int i = blockIdx.x * blockDim.x + threadIdx.x;
  if (i < n) p[i] = 0;
}

__global__ void k_deg(const int* __restrict__ dst, int* __restrict__ deg, int E) {
  int e = blockIdx.x * blockDim.x + threadIdx.x;
  if (e < E) atomicAdd(&deg[dst[e]], 1);
}

__global__ __launch_bounds__(1024) void k_scan1(const int* __restrict__ deg,
                                                int* __restrict__ rowptr,
                                                int* __restrict__ bsum, int N) {
  __shared__ int s[1024];
  int tid = threadIdx.x;
  int gid = blockIdx.x * 1024 + tid;
  int v = (gid < N) ? deg[gid] : 0;
  s[tid] = v;
  __syncthreads();
  for (int off = 1; off < 1024; off <<= 1) {
    int t = (tid >= off) ? s[tid - off] : 0;
    __syncthreads();
    s[tid] += t;
    __syncthreads();
  }
  if (gid < N) rowptr[gid] = s[tid] - v;        // block-local exclusive
  if (tid == 1023) bsum[blockIdx.x] = s[tid];
}

__global__ __launch_bounds__(256) void k_scan2(int* __restrict__ bsum, int NB) {
  __shared__ int s[256];
  int tid = threadIdx.x;
  int v = (tid < NB) ? bsum[tid] : 0;
  s[tid] = v;
  __syncthreads();
  for (int off = 1; off < 256; off <<= 1) {
    int t = (tid >= off) ? s[tid - off] : 0;
    __syncthreads();
    s[tid] += t;
    __syncthreads();
  }
  if (tid < NB) bsum[tid] = s[tid] - v;         // exclusive block offsets
}

__global__ void k_scan3(int* __restrict__ rowptr, const int* __restrict__ bsum,
                        int* __restrict__ cursor, int N, int E) {
  int gid = blockIdx.x * blockDim.x + threadIdx.x;
  if (gid < N) {
    int r = rowptr[gid] + bsum[gid >> 10];
    rowptr[gid] = r;
    cursor[gid] = r;
  }
  if (gid == 0) rowptr[N] = E;
}

__global__ void k_fill(const int* __restrict__ src, const int* __restrict__ dst,
                       int* __restrict__ cursor, int* __restrict__ eidx, int E) {
  int e = blockIdx.x * blockDim.x + threadIdx.x;
  if (e < E) {
    int pos = atomicAdd(&cursor[dst[e]], 1);
    eidx[pos] = src[e];
  }
}

// ---------------------------------------------------------------------------
// One wave per destination node: segment max -> exp-sum + weighted gather
// accumulation in registers (lane = feature, 8 heads unrolled), normalize,
// one coalesced 1KB store. No float atomics. q rows load as 2x float4.
// ---------------------------------------------------------------------------
__global__ __launch_bounds__(256) void k_out(const float* __restrict__ h,
                                             const float* __restrict__ q,
                                             const float* __restrict__ kk,
                                             const int* __restrict__ rowptr,
                                             const int* __restrict__ eidx,
                                             float* __restrict__ out, int N) {
  int wave = (int)((blockIdx.x * blockDim.x + threadIdx.x) >> 5);
  int lane = threadIdx.x & 31;
  if (wave >= N) return;
  int beg = rowptr[wave], end = rowptr[wave + 1];

  float kv[8];
  {
    float4 ka = *(const float4*)(kk + (size_t)wave * 8);
    float4 kb = *(const float4*)(kk + (size_t)wave * 8 + 4);
    kv[0] = ka.x; kv[1] = ka.y; kv[2] = ka.z; kv[3] = ka.w;
    kv[4] = kb.x; kv[5] = kb.y; kv[6] = kb.z; kv[7] = kb.w;
  }

  float mx[8];
#pragma unroll
  for (int hd = 0; hd < 8; ++hd) mx[hd] = -3.402823466e38f;

  // pass 1: lane-parallel running max of LeakyReLU(q[src]+k[dst])
  for (int e = beg + lane; e < end; e += 32) {
    int s = eidx[e];
    float4 qa = *(const float4*)(q + (size_t)s * 8);
    float4 qb = *(const float4*)(q + (size_t)s * 8 + 4);
    float qv[8] = {qa.x, qa.y, qa.z, qa.w, qb.x, qb.y, qb.z, qb.w};
#pragma unroll
    for (int hd = 0; hd < 8; ++hd) {
      float c = qv[hd] + kv[hd];
      c = (c > 0.f) ? c : 0.2f * c;
      mx[hd] = fmaxf(mx[hd], c);
    }
  }
#pragma unroll
  for (int hd = 0; hd < 8; ++hd) {
#pragma unroll
    for (int off = 16; off > 0; off >>= 1)
      mx[hd] = fmaxf(mx[hd], __shfl_xor(mx[hd], off, 32));
  }

  float acc[8], den[8];
#pragma unroll
  for (int hd = 0; hd < 8; ++hd) { acc[hd] = 0.f; den[hd] = 0.f; }

  // pass 2: un-normalized weighted sum + denominator (divide at the end)
  for (int e = beg; e < end; ++e) {
    int s = eidx[e];                             // uniform across wave
    const float* hr = h + (size_t)s * 256;
    float4 qa = *(const float4*)(q + (size_t)s * 8);
    float4 qb = *(const float4*)(q + (size_t)s * 8 + 4);
    float qv[8] = {qa.x, qa.y, qa.z, qa.w, qb.x, qb.y, qb.z, qb.w};
#pragma unroll
    for (int hd = 0; hd < 8; ++hd) {
      float c = qv[hd] + kv[hd];
      c = (c > 0.f) ? c : 0.2f * c;
      float ex = __expf(c - mx[hd]);
      den[hd] += ex;
      acc[hd] += ex * hr[hd * 32 + lane];
    }
  }

  float* orow = out + (size_t)wave * 256;
#pragma unroll
  for (int hd = 0; hd < 8; ++hd)
    orow[hd * 32 + lane] = (end > beg) ? acc[hd] / den[hd] : 0.f;
}

// ---------------------------------------------------------------------------
extern "C" void kernel_launch(void* const* d_in, const int* in_sizes, int n_in,
                              void* d_out, int out_size, void* d_ws, size_t ws_size,
                              hipStream_t stream) {
  const float* x   = (const float*)d_in[0];
  const int*   src = (const int*)d_in[1];
  const int*   dst = (const int*)d_in[2];
  const float* Wv  = (const float*)d_in[3];
  const float* bv  = (const float*)d_in[4];
  const float* Wq  = (const float*)d_in[5];
  const float* bq  = (const float*)d_in[6];
  const float* Wk  = (const float*)d_in[7];
  const float* bk  = (const float*)d_in[8];
  float* out = (float*)d_out;

  const int N = in_sizes[0] / 256;
  const int E = in_sizes[1];
  (void)n_in; (void)out_size; (void)ws_size;

  char* ws = (char*)d_ws;
  size_t off = 0;
  auto take = [&](size_t bytes) -> char* {
    char* p = ws + off;
    off += (bytes + 255) & ~(size_t)255;
    return p;
  };
  float*        h      = (float*)take((size_t)N * 256 * 4);
  float*        q      = (float*)take((size_t)N * 8 * 4);
  float*        kk     = (float*)take((size_t)N * 8 * 4);
  unsigned int* Wsw    = (unsigned int*)take((size_t)8 * 16 * 32 * 8 * 4);
  int*          deg    = (int*)take((size_t)N * 4);
  int*          rowptr = (int*)take(((size_t)N + 1) * 4);
  int*          cursor = (int*)take((size_t)N * 4);
  int*          bsum   = (int*)take((size_t)1024 * 4);
  int*          eidx   = (int*)take((size_t)E * 4);

  const int numTiles = (N + 15) / 16;
  const int NB       = (N + 1023) / 1024;       // scan blocks (<=256 for N<=256K)

  k_prep_w<<<dim3((8 * 16 * 32 * 8 + 255) / 256), dim3(256), 0, stream>>>(Wv, Wsw);
  k_gemm_h<<<dim3((numTiles * 32 + 255) / 256), dim3(256), 0, stream>>>(
      x, Wsw, bv, h, numTiles, N);
  k_qk<<<dim3((N * 32 + 255) / 256), dim3(256), 0, stream>>>(h, Wq, bq, Wk, bk, q, kk, N);

  k_zero_i32<<<dim3((N + 255) / 256), dim3(256), 0, stream>>>(deg, N);
  k_deg<<<dim3((E + 255) / 256), dim3(256), 0, stream>>>(dst, deg, E);
  k_scan1<<<dim3(NB), dim3(1024), 0, stream>>>(deg, rowptr, bsum, N);
  k_scan2<<<dim3(1), dim3(256), 0, stream>>>(bsum, NB);
  k_scan3<<<dim3((N + 255) / 256), dim3(256), 0, stream>>>(rowptr, bsum, cursor, N, E);
  k_fill<<<dim3((E + 255) / 256), dim3(256), 0, stream>>>(src, dst, cursor, eidx, E);

  k_out<<<dim3((N * 32 + 255) / 256), dim3(256), 0, stream>>>(h, q, kk, rowptr, eidx, out, N);
}
